// CausalSelfAttention_79104707658451
// MI455X (gfx1250) — matrix operand order repaired and checked by reading, not compiled, not run on hardware
//
#include <hip/hip_runtime.h>
#include <hip/hip_bf16.h>

// ---------------------------------------------------------------------------
// CDNA5 (gfx1250) bf16 WMMA building blocks
// ---------------------------------------------------------------------------
typedef __attribute__((ext_vector_type(16))) __bf16 v16bf;
typedef __attribute__((ext_vector_type(8)))  __bf16 v8bf;
typedef __attribute__((ext_vector_type(8)))  float  v8f;

// Fixed problem dims (reference: B=2, T=2048, E=1024, H=16, D=64)
constexpr int GN   = 1024;   // N of all projections (= E)
constexpr int GK   = 1024;   // K of all projections (= E)
constexpr int TLEN = 2048;

#if __has_builtin(__builtin_amdgcn_global_load_async_to_lds_b128) && \
    __has_builtin(__builtin_amdgcn_s_wait_asynccnt)
#define HAVE_ASYNC_LDS 1
#endif

#ifdef HAVE_ASYNC_LDS
// Builtin signature (from hipcc diagnostic): params are pointers to
// int __attribute__((vector_size(16))) in addrspace(1) / addrspace(3).
typedef int vs4i __attribute__((vector_size(4 * sizeof(int))));
typedef __attribute__((address_space(1))) vs4i gvs4i;
typedef __attribute__((address_space(3))) vs4i lvs4i;
#endif

// 16-byte global -> LDS copy; uses CDNA5 async-to-LDS (ASYNCcnt) when available
__device__ __forceinline__ void copy16_g2l(const void* g, void* l) {
#ifdef HAVE_ASYNC_LDS
  __builtin_amdgcn_global_load_async_to_lds_b128((gvs4i*)g, (lvs4i*)l, 0, 0);
#else
  *(uint4*)l = *(const uint4*)g;
#endif
}
__device__ __forceinline__ void async_fence() {
#ifdef HAVE_ASYNC_LDS
  __builtin_amdgcn_s_wait_asynccnt(0);
#endif
}

__device__ __forceinline__ __bf16 f2bf(float f) {
  union { float f; unsigned u; } a; a.f = f;
  unsigned r = a.u + 0x7FFFu + ((a.u >> 16) & 1u);   // round-to-nearest-even
  union { unsigned short s; __bf16 b; } o;
  o.s = (unsigned short)(r >> 16);
  return o.b;
}

// A fragment (16x32, 16-bit): lane m = lane&15, khalf = lane>>4.
// elems 0..7 -> K = khalf*8 + 0..7 ; elems 8..15 -> K = 16 + khalf*8 + 0..7
__device__ __forceinline__ v16bf load_frag_a(const __bf16* base, int ld) {
  const int lane = threadIdx.x & 31;
  const int row  = lane & 15;
  const int kh   = (lane >> 4) & 1;
  const __bf16* p = base + row * ld + kh * 8;
  union { v16bf v; v8bf h[2]; } u;
  u.h[0] = *(const v8bf*)(p);
  u.h[1] = *(const v8bf*)(p + 16);
  return u.v;
}

// B fragment (32x16, 16-bit): lane n = lane&15, elems 0..7 -> K = khalf*8 + 0..7 ; elems 8..15 -> K = 16 + khalf*8 + 0..7.
// Source tile stored transposed [n][k]: 16 K values/lane = two 16-byte runs, 32 bytes apart.
__device__ __forceinline__ v16bf load_frag_b(const __bf16* base, int ld) {
  const int lane = threadIdx.x & 31;
  const int col  = lane & 15;
  const int kb   = (lane >> 4) * 8;
  const __bf16* p = base + col * ld + kb;
  union { v16bf v; v8bf h[2]; } u;
  u.h[0] = *(const v8bf*)(p);
  u.h[1] = *(const v8bf*)(p + 16);
  return u.v;
}

__device__ __forceinline__ v8f wmma_bf16(v16bf a, v16bf b, v8f c) {
  return __builtin_amdgcn_wmma_f32_16x16x32_bf16(
      false, a, false, b, (short)0, c, false, false);
}

// ---------------------------------------------------------------------------
// GEMM: out = bf16(X) @ bf16(W) + bias.  X: M x 1024 fp32, W: 1024 x 1024 fp32.
// mode 0: out fp32 row-major (M x N)
// mode 1: out bf16 in (B, H=16, T=2048, D=64) layout
// Block tile 128x64, BK=64, 256 threads (8 waves; each wave 32x32 = 4 C-frags,
// 8 WMMAs per K step).  Global loads software-pipelined against WMMA.
// ---------------------------------------------------------------------------
__global__ __launch_bounds__(256) void proj_gemm_kernel(
    const float* __restrict__ X, const float* __restrict__ W,
    const float* __restrict__ bias, void* __restrict__ outp, int mode) {
  __shared__ __align__(16) __bf16 sA[128][64];   // X tile [m][k]   16KB
  __shared__ __align__(16) __bf16 sB[64][64];    // W tile [n][k]    8KB

  const int tid  = threadIdx.x;
  const int wv   = tid >> 5;
  const int lane = tid & 31;
  const int hf   = lane >> 4;
  const int nn   = lane & 15;
  const int m0   = blockIdx.x * 128;
  const int n0   = blockIdx.y * 64;
  const int wrow = (wv >> 1) * 32;
  const int wcol = (wv & 1) * 32;

  float4 ra[8], rb[4];

  auto load_tiles = [&](int k0) {
#pragma unroll
    for (int t = 0; t < 8; ++t) {            // A: 128x64 = 2048 float4 chunks
      const int c = tid + t * 256;
      const int row = c >> 4, kk = (c & 15) * 4;
      ra[t] = *(const float4*)(X + (size_t)(m0 + row) * GK + k0 + kk);
    }
#pragma unroll
    for (int t = 0; t < 4; ++t) {            // B: 64x64 = 1024 float4 chunks
      const int c = tid + t * 256;
      const int kk = c >> 4, n4 = (c & 15) * 4;
      rb[t] = *(const float4*)(W + (size_t)(k0 + kk) * GN + n0 + n4);
    }
  };
  auto store_tiles = [&]() {
#pragma unroll
    for (int t = 0; t < 8; ++t) {
      const int c = tid + t * 256;
      const int row = c >> 4, kk = (c & 15) * 4;
      sA[row][kk + 0] = f2bf(ra[t].x);
      sA[row][kk + 1] = f2bf(ra[t].y);
      sA[row][kk + 2] = f2bf(ra[t].z);
      sA[row][kk + 3] = f2bf(ra[t].w);
    }
#pragma unroll
    for (int t = 0; t < 4; ++t) {
      const int c = tid + t * 256;
      const int kk = c >> 4, n4 = (c & 15) * 4;
      sB[n4 + 0][kk] = f2bf(rb[t].x);
      sB[n4 + 1][kk] = f2bf(rb[t].y);
      sB[n4 + 2][kk] = f2bf(rb[t].z);
      sB[n4 + 3][kk] = f2bf(rb[t].w);
    }
  };

  v8f c00 = {0.f,0.f,0.f,0.f,0.f,0.f,0.f,0.f};
  v8f c01 = c00, c10 = c00, c11 = c00;

  load_tiles(0);
  for (int k0 = 0; k0 < GK; k0 += 64) {
    store_tiles();
    __syncthreads();
    if (k0 + 64 < GK) load_tiles(k0 + 64);   // prefetch next tile -> registers
#pragma unroll
    for (int ks = 0; ks < 64; ks += 32) {
      const v16bf a0 = load_frag_a(&sA[wrow][ks], 64);
      const v16bf a1 = load_frag_a(&sA[wrow + 16][ks], 64);
      const v16bf b0 = load_frag_b(&sB[wcol][ks], 64);
      const v16bf b1 = load_frag_b(&sB[wcol + 16][ks], 64);
      c00 = wmma_bf16(a0, b0, c00);
      c01 = wmma_bf16(a0, b1, c01);
      c10 = wmma_bf16(a1, b0, c10);
      c11 = wmma_bf16(a1, b1, c11);
    }
    __syncthreads();
  }

  const float bias0 = bias[n0 + wcol + nn];
  const float bias1 = bias[n0 + wcol + 16 + nn];
  const v8f* frag[2][2] = {{&c00, &c01}, {&c10, &c11}};
#pragma unroll
  for (int rf = 0; rf < 2; ++rf) {
#pragma unroll
    for (int cf = 0; cf < 2; ++cf) {
#pragma unroll
      for (int r = 0; r < 8; ++r) {
        const int m = m0 + wrow + rf * 16 + hf * 8 + r;
        const int n = n0 + wcol + cf * 16 + nn;
        const float v = (*frag[rf][cf])[r] + (cf ? bias1 : bias0);
        if (mode == 0) {
          ((float*)outp)[(size_t)m * GN + n] = v;
        } else {
          const int bb = m >> 11, t = m & (TLEN - 1);     // TLEN = 2048
          ((__bf16*)outp)[(((size_t)bb * 16 + (n >> 6)) * TLEN + t) * 64 +
                          (n & 63)] = f2bf(v);
        }
      }
    }
  }
}

// ---------------------------------------------------------------------------
// Flash attention (causal), Q/K/V bf16 in (B,H,2048,64).  Grid (T/64, H, B),
// 128 threads = 4 waves, each wave owns 16 query rows.  Output fp32 (B,T,E).
// ---------------------------------------------------------------------------
__global__ __launch_bounds__(128) void flash_attn_kernel(
    const __bf16* __restrict__ Q, const __bf16* __restrict__ K,
    const __bf16* __restrict__ V, float* __restrict__ O) {
  __shared__ __align__(16) __bf16 sQ[64][64];      // [qrow][d]
  __shared__ __align__(16) __bf16 sK[64][64];      // [key][d] == B^T for QK^T
  __shared__ __align__(16) __bf16 sV[64][64];      // transposed [d][key]
  __shared__ __align__(16) __bf16 sP[4][16][64];   // per-wave P [row][key]

  const int tid  = threadIdx.x;
  const int wv   = tid >> 5;
  const int lane = tid & 31;
  const int hf   = lane >> 4;
  const int nn   = lane & 15;
  const int bx   = blockIdx.x;
  const int q0   = bx * 64;
  const size_t head_off =
      ((size_t)blockIdx.z * 16 + blockIdx.y) * (size_t)TLEN * 64;
  const __bf16* Qg = Q + head_off;
  const __bf16* Kg = K + head_off;
  const __bf16* Vg = V + head_off;

  // Stage Q tile (async global->LDS when available)
#pragma unroll
  for (int it = 0; it < 4; ++it) {
    const int c = tid + it * 128;
    const int row = c >> 3, d0 = (c & 7) * 8;
    copy16_g2l(Qg + (size_t)(q0 + row) * 64 + d0, &sQ[row][d0]);
  }
  async_fence();
  __syncthreads();
  const v16bf qa0 = load_frag_a(&sQ[wv * 16][0], 64);
  const v16bf qa1 = load_frag_a(&sQ[wv * 16][32], 64);

  v8f ofrag[4];
  float mrow[8], lrow[8];
#pragma unroll
  for (int nt = 0; nt < 4; ++nt)
    ofrag[nt] = (v8f){0.f,0.f,0.f,0.f,0.f,0.f,0.f,0.f};
#pragma unroll
  for (int r = 0; r < 8; ++r) { mrow[r] = -3.0e38f; lrow[r] = 0.f; }

  for (int j = 0; j <= bx; ++j) {
    __syncthreads();   // previous iteration's reads of sK/sV done
    const int k0 = j * 64;
#pragma unroll
    for (int it = 0; it < 4; ++it) {
      const int c = tid + it * 128;
      const int row = c >> 3, d0 = (c & 7) * 8;
      copy16_g2l(Kg + (size_t)(k0 + row) * 64 + d0, &sK[row][d0]);
      union { uint4 u; __bf16 b[8]; } cvt;
      cvt.u = *(const uint4*)(Vg + (size_t)(k0 + row) * 64 + d0);
#pragma unroll
      for (int q = 0; q < 8; ++q) sV[d0 + q][row] = cvt.b[q];
    }
    async_fence();
    __syncthreads();

    // S = Q @ K^T  (per wave 16x64: 4 n-frags, D accumulated 2x32)
    v8f s[4];
#pragma unroll
    for (int nt = 0; nt < 4; ++nt) {
      v8f acc = {0.f,0.f,0.f,0.f,0.f,0.f,0.f,0.f};
      acc = wmma_bf16(qa0, load_frag_b(&sK[nt * 16][0], 64), acc);
      acc = wmma_bf16(qa1, load_frag_b(&sK[nt * 16][32], 64), acc);
      s[nt] = acc;
    }

    const bool diag = (j == bx);
#pragma unroll
    for (int nt = 0; nt < 4; ++nt) {
#pragma unroll
      for (int r = 0; r < 8; ++r) {
        float v = s[nt][r] * 0.125f;   // 1/sqrt(64)
        if (diag) {
          const int kloc = nt * 16 + nn;
          const int qloc = wv * 16 + hf * 8 + r;
          if (kloc > qloc) v = -3.0e38f;
        }
        s[nt][r] = v;
      }
    }

    // online softmax: rows live in 16-lane halves -> xor masks 1/2/4/8
    float mnew[8], alpha[8], rs[8];
#pragma unroll
    for (int r = 0; r < 8; ++r) {
      float v = fmaxf(fmaxf(s[0][r], s[1][r]), fmaxf(s[2][r], s[3][r]));
      v = fmaxf(v, __shfl_xor(v, 1, 32));
      v = fmaxf(v, __shfl_xor(v, 2, 32));
      v = fmaxf(v, __shfl_xor(v, 4, 32));
      v = fmaxf(v, __shfl_xor(v, 8, 32));
      mnew[r]  = fmaxf(mrow[r], v);
      alpha[r] = __expf(mrow[r] - mnew[r]);
      rs[r]    = 0.f;
    }

#pragma unroll
    for (int nt = 0; nt < 4; ++nt) {
#pragma unroll
      for (int r = 0; r < 8; ++r) {
        const float p = __expf(s[nt][r] - mnew[r]);
        rs[r] += p;
        sP[wv][hf * 8 + r][nt * 16 + nn] = f2bf(p);
      }
    }

#pragma unroll
    for (int r = 0; r < 8; ++r) {
      float v = rs[r];
      v += __shfl_xor(v, 1, 32);
      v += __shfl_xor(v, 2, 32);
      v += __shfl_xor(v, 4, 32);
      v += __shfl_xor(v, 8, 32);
      lrow[r] = lrow[r] * alpha[r] + v;
      mrow[r] = mnew[r];
    }

#pragma unroll
    for (int nt = 0; nt < 4; ++nt)
#pragma unroll
      for (int r = 0; r < 8; ++r)
        ofrag[nt][r] = ofrag[nt][r] * alpha[r];

    // O += P @ V
    const v16bf pa0 = load_frag_a(&sP[wv][0][0], 64);
    const v16bf pa1 = load_frag_a(&sP[wv][0][32], 64);
#pragma unroll
    for (int nt = 0; nt < 4; ++nt) {
      ofrag[nt] = wmma_bf16(pa0, load_frag_b(&sV[nt * 16][0], 64), ofrag[nt]);
      ofrag[nt] = wmma_bf16(pa1, load_frag_b(&sV[nt * 16][32], 64), ofrag[nt]);
    }
  }

  // normalize, write (B, T, E) fp32 for the output projection
#pragma unroll
  for (int nt = 0; nt < 4; ++nt) {
#pragma unroll
    for (int r = 0; r < 8; ++r) {
      const float v = ofrag[nt][r] / lrow[r];
      const int t = q0 + wv * 16 + hf * 8 + r;
      const int e = blockIdx.y * 64 + nt * 16 + nn;
      O[((size_t)blockIdx.z * TLEN + t) * 1024 + e] = v;
    }
  }
}

// ---------------------------------------------------------------------------
// Launch
// ---------------------------------------------------------------------------
extern "C" void kernel_launch(void* const* d_in, const int* in_sizes, int n_in,
                              void* d_out, int out_size, void* d_ws, size_t ws_size,
                              hipStream_t stream) {
  const float* x  = (const float*)d_in[0];
  const float* Wq = (const float*)d_in[1];
  const float* bq = (const float*)d_in[2];
  const float* Wk = (const float*)d_in[3];
  const float* bk = (const float*)d_in[4];
  const float* Wv = (const float*)d_in[5];
  const float* bv = (const float*)d_in[6];
  const float* Wo = (const float*)d_in[7];
  const float* bo = (const float*)d_in[8];

  const int E = 1024;
  const int T = 2048;
  const int B = in_sizes[0] / (T * E);   // 2
  const int M = B * T;                   // 4096

  char* ws = (char*)d_ws;
  __bf16* Qb = (__bf16*)(ws);
  __bf16* Kb = (__bf16*)(ws + (size_t)M * E * 2);
  __bf16* Vb = (__bf16*)(ws + (size_t)M * E * 4);
  float*  Ob = (float*)(ws + (size_t)M * E * 6);

  dim3 gp(M / 128, E / 64);
  proj_gemm_kernel<<<gp, 256, 0, stream>>>(x, Wq, bq, Qb, 1);
  proj_gemm_kernel<<<gp, 256, 0, stream>>>(x, Wk, bk, Kb, 1);
  proj_gemm_kernel<<<gp, 256, 0, stream>>>(x, Wv, bv, Vb, 1);

  dim3 ga(T / 64, 16, B);
  flash_attn_kernel<<<ga, 128, 0, stream>>>(Qb, Kb, Vb, Ob);

  proj_gemm_kernel<<<gp, 256, 0, stream>>>(Ob, Wo, bo, d_out, 0);
}
